// NormalEstimation_53051436040783
// MI455X (gfx1250) — matrix operand order repaired and checked by reading, not compiled, not run on hardware
//
#include <hip/hip_runtime.h>
#include <math.h>

// CDNA5 (gfx1250) wave32 WMMA implementation of the normal-estimation MLP.
// One wave (32 lanes) per point: 16 v_wmma_f32_16x16x32_f16 ops per point.

typedef __attribute__((ext_vector_type(16))) _Float16 v16h;
typedef __attribute__((ext_vector_type(8)))  float    v8f;
typedef __attribute__((ext_vector_type(4)))  unsigned u32x4;

union FragU { u32x4 u[2]; v16h h; };

__device__ __forceinline__ void lds_fence() { asm volatile("" ::: "memory"); }

// wave32 xor-shuffle via ds_bpermute (index wraps mod 32 on wave32)
__device__ __forceinline__ float shx(float v, int m) {
  int lane = (int)(threadIdx.x & 31u);
  int idx = ((lane ^ m) & 31) << 2;
  return __int_as_float(__builtin_amdgcn_ds_bpermute(idx, __float_as_int(v)));
}

__device__ __forceinline__ unsigned pk2(float a, float b) {
  union { _Float16 h[2]; unsigned u; } x;
  x.h[0] = (_Float16)a; x.h[1] = (_Float16)b;
  return x.u;
}

// A fragment (16x32 f16, MxK): lane holds row M=16*mt+(lane&15);
// halves 0..7 -> K=8g..8g+7 ; halves 8..15 -> K=16+8g..16+8g+7  (g = lane>>4)
__device__ __forceinline__ v16h frag_a(const _Float16* base, int mt, int lane) {
  int gg = lane >> 4;
  int row = 16 * mt + (lane & 15);
  const char* p = (const char*)base + row * 64;   // 32 halves per row
  FragU f;
  f.u[0] = *(const u32x4*)(p + 16 * gg);
  f.u[1] = *(const u32x4*)(p + 32 + 16 * gg);
  return f.h;
}

// A fragment whose every row equals the 32-value vector g[] (broadcast rows)
__device__ __forceinline__ v16h frag_g(const _Float16* g, int lane) {
  int gg = lane >> 4;
  const char* p = (const char*)g;
  FragU f;
  f.u[0] = *(const u32x4*)(p + 16 * gg);
  f.u[1] = *(const u32x4*)(p + 32 + 16 * gg);
  return f.h;
}

// B fragment (32x16 f16, KxN) from transposed weights WT[N][K] (row-major):
// lane holds column N=16*nt+(lane&15); halves 0..7 -> K = kOff + 8g + h ; halves 8..15 -> K = kOff + 16 + 8g + (h-8)
__device__ __forceinline__ v16h frag_b(const _Float16* wt, int rowBytes, int nt,
                                       int kOffHalves, int lane) {
  int gg = lane >> 4;
  int nrow = 16 * nt + (lane & 15);
  const char* p = (const char*)wt + nrow * rowBytes + kOffHalves * 2 + 16 * gg;
  FragU f;
  f.u[0] = *(const u32x4*)(p);
  f.u[1] = *(const u32x4*)(p + 32);
  return f.h;
}

__device__ __forceinline__ v8f wmma16(v16h a, v16h b, v8f c) {
  return __builtin_amdgcn_wmma_f32_16x16x32_f16(false, a, false, b, (short)0, c,
                                                false, false);
}

// D tile (v8f): element r -> row M=16*mt+8g+r, col N=16*nt+(lane&15).
// Apply bias+ReLU, store f16 into activation buffer A[32][32]; optionally
// accumulate column sum (for mean pooling).
__device__ __forceinline__ void store_tile_relu(_Float16* A, v8f d, int mt, int nt,
                                                const float* bias, int lane,
                                                float* csum) {
  int gg = lane >> 4;
  int N = 16 * nt + (lane & 15);
  float b = bias[N];
#pragma unroll
  for (int r = 0; r < 8; ++r) {
    float v = fmaxf(d[r] + b, 0.f);
    if (csum) *csum += v;
    int M = 16 * mt + 8 * gg + r;
    A[M * 32 + N] = (_Float16)v;
  }
}

__global__ __launch_bounds__(256)
void NormalEstWMMA_kernel(const float* __restrict__ old_w,
                          const float* __restrict__ pos,
                          const float* __restrict__ normals,
                          const int*   __restrict__ dense,
                          const float* __restrict__ stddev,
                          const float* __restrict__ W1, const float* __restrict__ B1,
                          const float* __restrict__ W2, const float* __restrict__ B2,
                          const float* __restrict__ W3, const float* __restrict__ B3,
                          const float* __restrict__ W4, const float* __restrict__ B4,
                          float* __restrict__ nout, float* __restrict__ wout, int N) {
  // block-shared transposed f16 weights + f32 biases
  __shared__ alignas(16) _Float16 wt1[32 * 32];   // WT1[n][k], k<6 valid, rest 0
  __shared__ alignas(16) _Float16 wt2[32 * 32];   // WT2[n][k]
  __shared__ alignas(16) _Float16 wt3[32 * 64];   // WT3[n][k], k in [0,64)
  __shared__ float sb1[32], sb2[32], sb3[32], sW4[32];
  // per-wave scratch
  __shared__ alignas(16) _Float16 abuf[8][32 * 32];
  __shared__ alignas(16) _Float16 gbuf[8][32];
  __shared__ float wbuf[8][32];

  int tid = (int)threadIdx.x;
  for (int i = tid; i < 32 * 32; i += 256) {
    int nn = i >> 5, kk = i & 31;
    wt1[i] = (_Float16)((kk < 6) ? W1[kk * 32 + nn] : 0.f);
    wt2[i] = (_Float16)W2[kk * 32 + nn];
  }
  for (int i = tid; i < 32 * 64; i += 256) {
    int nn = i >> 6, kk = i & 63;
    wt3[i] = (_Float16)W3[kk * 32 + nn];
  }
  if (tid < 32) { sb1[tid] = B1[tid]; sb2[tid] = B2[tid]; sb3[tid] = B3[tid]; sW4[tid] = W4[tid]; }
  __syncthreads();

  int lane = tid & 31;
  int wv = tid >> 5;
  int n = (int)blockIdx.x * 8 + wv;
  if (n >= N) return;

  _Float16* A = abuf[wv];
  _Float16* G = gbuf[wv];
  float*    WB = wbuf[wv];

  // ---- features: lane = neighbor index k ----
  int j = dense[n * 32 + lane];
  float px = pos[3 * n + 0], py = pos[3 * n + 1], pz = pos[3 * n + 2];
  float dx = pos[3 * j + 0] - px, dy = pos[3 * j + 1] - py, dz = pos[3 * j + 2] - pz;
  float inv = 1.f / stddev[0];
  float sx = dx * inv, sy = dy * inv, sz = dz * inv;
  float nx = normals[3 * n + 0], ny = normals[3 * n + 1], nz = normals[3 * n + 2];
  float proj = nx * sx + ny * sy + nz * sz;
  float dist = sqrtf(sx * sx + sy * sy + sz * sz);
  float ow = old_w[n * 32 + lane];

  u32x4 r0 = { pk2(sx, sy), pk2(sz, proj), pk2(dist, ow), 0u };
  u32x4 z4 = { 0u, 0u, 0u, 0u };
  u32x4* rowp = (u32x4*)(A + lane * 32);
  rowp[0] = r0; rowp[1] = z4; rowp[2] = z4; rowp[3] = z4;
  lds_fence();

  v8f zf = {0.f, 0.f, 0.f, 0.f, 0.f, 0.f, 0.f, 0.f};

  // ---- layer 1: h1 = relu(f @ W1 + b1)   [32x6(pad32)] @ [32x32] ----
  v16h a0 = frag_a(A, 0, lane), a1 = frag_a(A, 1, lane);
  v16h b0 = frag_b(wt1, 64, 0, 0, lane), b1f = frag_b(wt1, 64, 1, 0, lane);
  v8f d00 = wmma16(a0, b0, zf), d01 = wmma16(a0, b1f, zf);
  v8f d10 = wmma16(a1, b0, zf), d11 = wmma16(a1, b1f, zf);
  lds_fence();
  store_tile_relu(A, d00, 0, 0, sb1, lane, nullptr);
  store_tile_relu(A, d01, 0, 1, sb1, lane, nullptr);
  store_tile_relu(A, d10, 1, 0, sb1, lane, nullptr);
  store_tile_relu(A, d11, 1, 1, sb1, lane, nullptr);
  lds_fence();

  // ---- layer 2: h = relu(h1 @ W2 + b2), plus column-mean g ----
  a0 = frag_a(A, 0, lane); a1 = frag_a(A, 1, lane);
  b0 = frag_b(wt2, 64, 0, 0, lane); b1f = frag_b(wt2, 64, 1, 0, lane);
  d00 = wmma16(a0, b0, zf); d01 = wmma16(a0, b1f, zf);
  d10 = wmma16(a1, b0, zf); d11 = wmma16(a1, b1f, zf);
  lds_fence();
  float cs0 = 0.f, cs1 = 0.f;
  store_tile_relu(A, d00, 0, 0, sb2, lane, &cs0);
  store_tile_relu(A, d01, 0, 1, sb2, lane, &cs1);
  store_tile_relu(A, d10, 1, 0, sb2, lane, &cs0);
  store_tile_relu(A, d11, 1, 1, sb2, lane, &cs1);
  cs0 += shx(cs0, 16); cs1 += shx(cs1, 16);           // fold the two lane halves
  float g0 = cs0 * (1.f / 32.f), g1 = cs1 * (1.f / 32.f);
  if (lane < 16) { G[lane] = (_Float16)g0; G[16 + lane] = (_Float16)g1; }
  lds_fence();

  // ---- layer 3: h3 = relu([h, g] @ W3 + b3), K = 64 in two 32-slices ----
  a0 = frag_a(A, 0, lane); a1 = frag_a(A, 1, lane);
  v16h ag = frag_g(G, lane);                          // row-constant A slice
  v16h e00 = frag_b(wt3, 128, 0, 0,  lane), e01 = frag_b(wt3, 128, 1, 0,  lane);
  v16h e10 = frag_b(wt3, 128, 0, 32, lane), e11 = frag_b(wt3, 128, 1, 32, lane);
  d00 = wmma16(a0, e00, zf); d00 = wmma16(ag, e10, d00);
  d01 = wmma16(a0, e01, zf); d01 = wmma16(ag, e11, d01);
  d10 = wmma16(a1, e00, zf); d10 = wmma16(ag, e10, d10);
  d11 = wmma16(a1, e01, zf); d11 = wmma16(ag, e11, d11);

  // ---- layer 4: weights = sigmoid(h3 @ W4 + b4)  (VALU + lane reduce) ----
  int gg = lane >> 4;
  int N0 = lane & 15, N1 = 16 + (lane & 15);
  float w40 = sW4[N0], w41 = sW4[N1];
  float b3a = sb3[N0], b3b = sb3[N1];
  float acc0[8], acc1[8];
#pragma unroll
  for (int r = 0; r < 8; ++r) {
    float h0a = fmaxf(d00[r] + b3a, 0.f);
    float h0b = fmaxf(d01[r] + b3b, 0.f);
    acc0[r] = h0a * w40 + h0b * w41;                  // rows 8g+r
    float h1a = fmaxf(d10[r] + b3a, 0.f);
    float h1b = fmaxf(d11[r] + b3b, 0.f);
    acc1[r] = h1a * w40 + h1b * w41;                  // rows 16+8g+r
  }
#pragma unroll
  for (int m = 1; m <= 8; m <<= 1) {
#pragma unroll
    for (int r = 0; r < 8; ++r) { acc0[r] += shx(acc0[r], m); acc1[r] += shx(acc1[r], m); }
  }
  float bb4 = B4[0];
  if ((lane & 15) == 0) {                             // lanes 0 and 16
#pragma unroll
    for (int r = 0; r < 8; ++r) {
      WB[8 * gg + r]      = 1.f / (1.f + __expf(-(acc0[r] + bb4)));
      WB[16 + 8 * gg + r] = 1.f / (1.f + __expf(-(acc1[r] + bb4)));
    }
  }
  lds_fence();
  float wk = WB[lane];
  wout[(size_t)n * 32 + lane] = wk;

  // ---- weighted covariance of un-normalized d, then 3x3 eigensolve ----
  float cxx = wk * dx * dx, cxy = wk * dx * dy, cxz = wk * dx * dz;
  float cyy = wk * dy * dy, cyz = wk * dy * dz, czz = wk * dz * dz;
#pragma unroll
  for (int m = 1; m <= 16; m <<= 1) {
    cxx += shx(cxx, m); cxy += shx(cxy, m); cxz += shx(cxz, m);
    cyy += shx(cyy, m); cyz += shx(cyz, m); czz += shx(czz, m);
  }
  if (lane == 0) {
    float q  = (cxx + cyy + czz) * (1.f / 3.f);
    float p1 = cxy * cxy + cxz * cxz + cyz * cyz;
    float f00 = cxx - q, f11 = cyy - q, f22 = czz - q;
    float p2 = f00 * f00 + f11 * f11 + f22 * f22 + 2.f * p1;
    float p  = sqrtf(p2 * (1.f / 6.f));
    float lmin = q;
    if (p > 1e-20f) {
      float ip = 1.f / p;
      float t00 = f00 * ip, t11 = f11 * ip, t22 = f22 * ip;
      float t01 = cxy * ip, t02 = cxz * ip, t12 = cyz * ip;
      float detB = t00 * (t11 * t22 - t12 * t12)
                 - t01 * (t01 * t22 - t12 * t02)
                 + t02 * (t01 * t12 - t11 * t02);
      float rr  = fminf(1.f, fmaxf(-1.f, 0.5f * detB));
      float phi = acosf(rr) * (1.f / 3.f);
      lmin = q + 2.f * p * cosf(phi + 2.0943951023931953f);   // smallest eigval (PSD)
    }
    float m00 = cxx - lmin, m11 = cyy - lmin, m22 = czz - lmin;
    // eigenvector = largest cross product of rows of (C - lmin*I)
    float v1x = cxy * cyz - cxz * m11, v1y = cxz * cxy - m00 * cyz, v1z = m00 * m11 - cxy * cxy;
    float v2x = cxy * m22 - cxz * cyz, v2y = cxz * cxz - m00 * m22, v2z = m00 * cyz - cxy * cxz;
    float v3x = m11 * m22 - cyz * cyz, v3y = cyz * cxz - cxy * m22, v3z = cxy * cyz - m11 * cxz;
    float q1 = v1x * v1x + v1y * v1y + v1z * v1z;
    float q2 = v2x * v2x + v2y * v2y + v2z * v2z;
    float q3 = v3x * v3x + v3y * v3y + v3z * v3z;
    float vx = v1x, vy = v1y, vz = v1z, nb = q1;
    if (q2 > nb) { vx = v2x; vy = v2y; vz = v2z; nb = q2; }
    if (q3 > nb) { vx = v3x; vy = v3y; vz = v3z; nb = q3; }
    if (nb < 1e-30f) { vx = 1.f; vy = 0.f; vz = 0.f; nb = 1.f; }
    float inb = rsqrtf(nb);
    nout[(size_t)n * 3 + 0] = vx * inb;
    nout[(size_t)n * 3 + 1] = vy * inb;
    nout[(size_t)n * 3 + 2] = vz * inb;
  }
}

extern "C" void kernel_launch(void* const* d_in, const int* in_sizes, int n_in,
                              void* d_out, int out_size, void* d_ws, size_t ws_size,
                              hipStream_t stream) {
  (void)n_in; (void)out_size; (void)d_ws; (void)ws_size;
  const float* old_w   = (const float*)d_in[0];
  const float* pos     = (const float*)d_in[1];
  // d_in[2] = batch (unused)
  const float* normals = (const float*)d_in[3];
  // d_in[4] = edge_idx_l (unused; dense_l is the indexed form)
  const int*   dense   = (const int*)d_in[5];
  const float* stddev  = (const float*)d_in[6];
  const float* W1 = (const float*)d_in[7];  const float* B1 = (const float*)d_in[8];
  const float* W2 = (const float*)d_in[9];  const float* B2 = (const float*)d_in[10];
  const float* W3 = (const float*)d_in[11]; const float* B3 = (const float*)d_in[12];
  const float* W4 = (const float*)d_in[13]; const float* B4 = (const float*)d_in[14];

  int N = in_sizes[1] / 3;                 // pos is N x 3
  float* nout = (float*)d_out;             // new_normals: N*3 floats
  float* wout = nout + (size_t)N * 3;      // weights: N*32 floats

  dim3 block(256);                         // 8 wave32 waves -> 8 points per block
  dim3 grid((N + 7) / 8);
  NormalEstWMMA_kernel<<<grid, block, 0, stream>>>(
      old_w, pos, normals, dense, stddev,
      W1, B1, W2, B2, W3, B3, W4, B4, nout, wout, N);
}
